// RNN_15607911154124
// MI455X (gfx1250) — matrix operand order repaired and checked by reading, not compiled, not run on hardware
//
#include <hip/hip_runtime.h>
#include <hip/hip_bf16.h>
#include <math.h>

typedef __attribute__((ext_vector_type(16))) __bf16 v16bf;
typedef __attribute__((ext_vector_type(8)))  __bf16 v8bf;
typedef __attribute__((ext_vector_type(8)))  float  v8f;
typedef __attribute__((ext_vector_type(4)))  unsigned int v4u;

#define V_    32000
#define E_    512
#define H_    1024
#define N_    32
#define T_    256
#define G3H   3072     // 3*H
#define NCHUNK 250     // 32000 / 128
#define CHUNK  128

// ---------------------------------------------------------------------------
// WMMA tile loaders (wave32, 16x16x32 bf16). Both A and B come from row-major
// [row][K] arrays because every GEMM here is X @ W^T (lane's B column == W row).
//
// A-matrix 16x32 bf16 layout (ISA 7.12.2): lane L holds row M=L%16,
//   element e -> K = (e%8) + 8*(L/16) + 16*(e/8)   => two 16B runs, 32B apart
// B-matrix 32x16 bf16 layout: lane L holds column N=L%16,
//   element e -> K = (e%8) + 8*(L/16) + 16*(e/8)   => two 16B runs, 32B apart
// ---------------------------------------------------------------------------
__device__ __forceinline__ v16bf load_tile_A(const __bf16* __restrict__ base, int ld,
                                             int row_base, int k_base, int lane) {
    int r    = row_base + (lane & 15);
    int half = lane >> 4;
    const __bf16* p = base + (size_t)r * ld + k_base + half * 8;
    v8bf lo = *(const v8bf*)(p);        // K = kb + 8*half + 0..7
    v8bf hi = *(const v8bf*)(p + 16);   // K = kb + 16 + 8*half + 0..7
    return __builtin_shufflevector(lo, hi, 0,1,2,3,4,5,6,7,8,9,10,11,12,13,14,15);
}

// A-tile from LDS-staged row-major panel
__device__ __forceinline__ v16bf load_tile_A_lds(const __bf16* base, int ld,
                                                 int row_base, int k_base, int lane) {
    int r    = row_base + (lane & 15);
    int half = lane >> 4;
    const __bf16* p = base + r * ld + k_base + half * 8;
    v8bf lo = *(const v8bf*)(p);
    v8bf hi = *(const v8bf*)(p + 16);
    return __builtin_shufflevector(lo, hi, 0,1,2,3,4,5,6,7,8,9,10,11,12,13,14,15);
}

// B tile: same K order as A -> two 16B runs per lane, 32B apart.
__device__ __forceinline__ v16bf load_tile_B(const __bf16* __restrict__ base, int ld,
                                             int row_base, int k_base, int lane) {
    int r    = row_base + (lane & 15);
    int half = lane >> 4;
    const __bf16* p = base + (size_t)r * ld + k_base + half * 8;
    v8bf lo = *(const v8bf*)(p);
    v8bf hi = *(const v8bf*)(p + 16);
    return __builtin_shufflevector(lo, hi, 0,1,2,3,4,5,6,7,8,9,10,11,12,13,14,15);
}

__device__ __forceinline__ v8f wmma_bf16(v16bf a, v16bf b, v8f c) {
    return __builtin_amdgcn_wmma_f32_16x16x32_bf16(false, a, false, b, (short)0, c, false, false);
}

// ---------------------------------------------------------------------------
// One-time prep kernels
// ---------------------------------------------------------------------------
__global__ void k_cvt_bf16(const float* __restrict__ src, __bf16* __restrict__ dst, int n) {
    int i = blockIdx.x * blockDim.x + threadIdx.x;
    if (i < n) dst[i] = (__bf16)src[i];
}

// gather embeddings for all timesteps 0..T-2 and convert to bf16: x_all[t][m][e]
__global__ void k_embed_gather(const float* __restrict__ embed_w,
                               const int* __restrict__ words,
                               __bf16* __restrict__ x_all) {
    int t = blockIdx.x >> 5;   // 0..254
    int m = blockIdx.x & 31;   // 0..31
    int w = words[m * T_ + t];
    const float* src = embed_w + (size_t)w * E_;
    __bf16* dst = x_all + ((size_t)t * N_ + m) * E_;
    for (int i = threadIdx.x; i < E_; i += blockDim.x) dst[i] = (__bf16)src[i];
}

__global__ void k_init(const float* __restrict__ hidden,
                       float* __restrict__ h, __bf16* __restrict__ h_bf,
                       float* __restrict__ acc) {
    int i = blockIdx.x * blockDim.x + threadIdx.x;
    if (i < N_ * H_) { float v = hidden[i]; h[i] = v; h_bf[i] = (__bf16)v; }
    if (i < N_) acc[i] = 0.f;
}

// ---------------------------------------------------------------------------
// GRU gate matmuls: gi = x @ W_ih^T  (M=32,K=512,N=3072)
//                   gh = h @ W_hh^T  (M=32,K=1024,N=3072)
// grid 24 x (256 threads = 8 waves); one N-tile (16 cols) per wave, 2 M-tiles.
// ---------------------------------------------------------------------------
__global__ void k_gru_mm(const __bf16* __restrict__ x_all, int t,
                         const __bf16* __restrict__ h_bf,
                         const __bf16* __restrict__ w_ih_bf,
                         const __bf16* __restrict__ w_hh_bf,
                         float* __restrict__ gi, float* __restrict__ gh) {
    const int lane = threadIdx.x & 31;
    const int wave = threadIdx.x >> 5;
    const int nb   = (blockIdx.x * 8 + wave) * 16;
    const int half = lane >> 4;
    const int col  = nb + (lane & 15);
    const __bf16* x = x_all + (size_t)t * (N_ * E_);

    #pragma unroll
    for (int mt = 0; mt < 2; ++mt) {
        v8f ci = {};
        #pragma unroll 4
        for (int k = 0; k < E_; k += 32) {
            v16bf a = load_tile_A(x, E_, mt * 16, k, lane);
            v16bf b = load_tile_B(w_ih_bf, E_, nb, k, lane);
            ci = wmma_bf16(a, b, ci);
        }
        v8f ch = {};
        #pragma unroll 4
        for (int k = 0; k < H_; k += 32) {
            v16bf a = load_tile_A(h_bf, H_, mt * 16, k, lane);
            v16bf b = load_tile_B(w_hh_bf, H_, nb, k, lane);
            ch = wmma_bf16(a, b, ch);
        }
        #pragma unroll
        for (int e = 0; e < 8; ++e) {
            int m = mt * 16 + half * 8 + e;   // C layout: M = e + 8*half (+16*mt)
            gi[m * G3H + col] = ci[e];
            gh[m * G3H + col] = ch[e];
        }
    }
}

// ---------------------------------------------------------------------------
// GRU elementwise gates in fp32; updates h and its bf16 mirror.
// ---------------------------------------------------------------------------
__global__ void k_gates(const float* __restrict__ gi, const float* __restrict__ gh,
                        const float* __restrict__ b_ih, const float* __restrict__ b_hh,
                        float* __restrict__ h, __bf16* __restrict__ h_bf) {
    int idx = blockIdx.x * blockDim.x + threadIdx.x;
    if (idx >= N_ * H_) return;
    int m = idx >> 10;
    int j = idx & (H_ - 1);
    const float* gim = gi + m * G3H;
    const float* ghm = gh + m * G3H;
    float ir = gim[j]          + b_ih[j];
    float iz = gim[H_ + j]     + b_ih[H_ + j];
    float in_= gim[2*H_ + j]   + b_ih[2*H_ + j];
    float hr = ghm[j]          + b_hh[j];
    float hz = ghm[H_ + j]     + b_hh[H_ + j];
    float hn = ghm[2*H_ + j]   + b_hh[2*H_ + j];
    float r  = 1.f / (1.f + expf(-(ir + hr)));
    float z  = 1.f / (1.f + expf(-(iz + hz)));
    float nn = tanhf(in_ + r * hn);
    float hv = (1.f - z) * nn + z * h[idx];
    h[idx]    = hv;
    h_bf[idx] = (__bf16)hv;
}

// ---------------------------------------------------------------------------
// Emit: logits chunk of 128 vocab per block (8 waves x 16 cols), online
// log-sum-exp stats per row + in-flight target-logit extraction.
// grid 250 x 256.  h panel (32x1024 bf16 = 64KB) is staged in LDS once per
// block so the 8 waves' A reads hit LDS instead of re-pulling from cache.
// ---------------------------------------------------------------------------
__global__ void k_emit(const __bf16* __restrict__ h_bf,
                       const __bf16* __restrict__ emit_w_bf,
                       const float* __restrict__ emit_b,
                       const int* __restrict__ words, int t,
                       float* __restrict__ lse_m, float* __restrict__ lse_s,
                       float* __restrict__ tgt) {
    __shared__ __align__(32) __bf16 hA[N_ * H_];   // 64 KB staged A panel
    __shared__ float smax[8][32];
    __shared__ float ssum[8][32];
    const int lane = threadIdx.x & 31;
    const int wave = threadIdx.x >> 5;
    const int half = lane >> 4;
    const int nb   = blockIdx.x * CHUNK + wave * 16;

    // cooperative 64KB stage: 256 threads x 16 x 16B vector copies
    {
        const v4u* src = (const v4u*)h_bf;
        v4u* dst = (v4u*)hA;
        #pragma unroll
        for (int i = 0; i < (N_ * H_ * 2 / 16) / 256; ++i)
            dst[threadIdx.x + i * 256] = src[threadIdx.x + i * 256];
    }
    __syncthreads();

    v8f c0 = {}, c1 = {};
    #pragma unroll 2
    for (int k = 0; k < H_; k += 32) {
        v16bf a0 = load_tile_A_lds(hA, H_, 0, k, lane);
        v16bf a1 = load_tile_A_lds(hA, H_, 16, k, lane);
        v16bf b  = load_tile_B(emit_w_bf, H_, nb, k, lane);
        c0 = wmma_bf16(a0, b, c0);
        c1 = wmma_bf16(a1, b, c1);
    }

    const int n = nb + (lane & 15);
    const float bias = emit_b[n];
    float v0[8], v1[8], m0[8], m1[8], s0[8], s1[8];
    #pragma unroll
    for (int e = 0; e < 8; ++e) {
        v0[e] = c0[e] + bias;  v1[e] = c1[e] + bias;
        m0[e] = v0[e];         m1[e] = v1[e];
    }
    // row-max across the wave's 16 columns (xor masks 1..8 stay inside a half)
    #pragma unroll
    for (int xm = 1; xm < 16; xm <<= 1) {
        #pragma unroll
        for (int e = 0; e < 8; ++e) {
            m0[e] = fmaxf(m0[e], __shfl_xor(m0[e], xm, 32));
            m1[e] = fmaxf(m1[e], __shfl_xor(m1[e], xm, 32));
        }
    }
    #pragma unroll
    for (int e = 0; e < 8; ++e) {
        s0[e] = expf(v0[e] - m0[e]);
        s1[e] = expf(v1[e] - m1[e]);
    }
    #pragma unroll
    for (int xm = 1; xm < 16; xm <<= 1) {
        #pragma unroll
        for (int e = 0; e < 8; ++e) {
            s0[e] += __shfl_xor(s0[e], xm, 32);
            s1[e] += __shfl_xor(s1[e], xm, 32);
        }
    }
    if ((lane & 15) == 0) {
        #pragma unroll
        for (int e = 0; e < 8; ++e) {
            smax[wave][half * 8 + e]      = m0[e];
            ssum[wave][half * 8 + e]      = s0[e];
            smax[wave][16 + half * 8 + e] = m1[e];
            ssum[wave][16 + half * 8 + e] = s1[e];
        }
    }
    // target logit extraction (exactly one lane in the grid matches each row)
    #pragma unroll
    for (int e = 0; e < 8; ++e) {
        int r0 = half * 8 + e;
        if (words[r0 * T_ + t + 1] == n) tgt[r0] = v0[e];
        int r1 = 16 + half * 8 + e;
        if (words[r1 * T_ + t + 1] == n) tgt[r1] = v1[e];
    }
    __syncthreads();
    if (threadIdx.x < 32) {
        int row = threadIdx.x;
        float M = smax[0][row], S = ssum[0][row];
        #pragma unroll
        for (int w = 1; w < 8; ++w) {
            float m = smax[w][row], s = ssum[w][row];
            float nm = fmaxf(M, m);
            S = S * expf(M - nm) + s * expf(m - nm);
            M = nm;
        }
        lse_m[blockIdx.x * 32 + row] = M;
        lse_s[blockIdx.x * 32 + row] = S;
    }
}

// merge 250 chunk (max,sum) pairs per row, accumulate log-likelihood
__global__ void k_reduce(const float* __restrict__ lse_m, const float* __restrict__ lse_s,
                         const float* __restrict__ tgt, float* __restrict__ acc) {
    int nrow = threadIdx.x;
    if (nrow >= N_) return;
    float M = -3.402823e38f, S = 0.f;
    for (int c = 0; c < NCHUNK; ++c) {
        float m = lse_m[c * 32 + nrow], s = lse_s[c * 32 + nrow];
        float nm = fmaxf(M, m);
        S = S * expf(M - nm) + s * expf(m - nm);
        M = nm;
    }
    acc[nrow] += tgt[nrow] - (logf(S) + M);
}

__global__ void k_out(const float* __restrict__ acc, const float* __restrict__ h,
                      float* __restrict__ out) {
    int i = blockIdx.x * blockDim.x + threadIdx.x;
    if (i >= N_ + N_ * H_) return;
    out[i] = (i < N_) ? acc[i] : h[i - N_];
}

// ---------------------------------------------------------------------------
extern "C" void kernel_launch(void* const* d_in, const int* in_sizes, int n_in,
                              void* d_out, int out_size, void* d_ws, size_t ws_size,
                              hipStream_t stream) {
    const int*   words   = (const int*)  d_in[0];
    const float* hidden  = (const float*)d_in[1];
    const float* embed_w = (const float*)d_in[2];
    const float* w_ih    = (const float*)d_in[3];
    const float* w_hh    = (const float*)d_in[4];
    const float* b_ih    = (const float*)d_in[5];
    const float* b_hh    = (const float*)d_in[6];
    const float* emit_w  = (const float*)d_in[7];
    const float* emit_b  = (const float*)d_in[8];
    float* out = (float*)d_out;

    char* ws = (char*)d_ws;
    size_t off = 0;
    auto alloc = [&](size_t bytes) {
        char* p = ws + off;
        off = (off + bytes + 255) & ~(size_t)255;
        return p;
    };
    __bf16* emit_w_bf = (__bf16*)alloc((size_t)V_ * H_ * 2);        // 64 MB
    __bf16* w_ih_bf   = (__bf16*)alloc((size_t)G3H * E_ * 2);       //  3 MB
    __bf16* w_hh_bf   = (__bf16*)alloc((size_t)G3H * H_ * 2);       //  6 MB
    __bf16* x_all     = (__bf16*)alloc((size_t)(T_ - 1) * N_ * E_ * 2); // 8 MB
    float*  h         = (float*) alloc((size_t)N_ * H_ * 4);
    __bf16* h_bf      = (__bf16*)alloc((size_t)N_ * H_ * 2);
    float*  gi        = (float*) alloc((size_t)N_ * G3H * 4);
    float*  gh        = (float*) alloc((size_t)N_ * G3H * 4);
    float*  lse_m     = (float*) alloc((size_t)NCHUNK * 32 * 4);
    float*  lse_s     = (float*) alloc((size_t)NCHUNK * 32 * 4);
    float*  tgt       = (float*) alloc(32 * 4);
    float*  acc       = (float*) alloc(32 * 4);

    // one-time prep: bf16 weight conversion (L2-resident afterwards), embed gather
    k_cvt_bf16<<<(V_ * H_ + 255) / 256, 256, 0, stream>>>(emit_w, emit_w_bf, V_ * H_);
    k_cvt_bf16<<<(G3H * E_ + 255) / 256, 256, 0, stream>>>(w_ih, w_ih_bf, G3H * E_);
    k_cvt_bf16<<<(G3H * H_ + 255) / 256, 256, 0, stream>>>(w_hh, w_hh_bf, G3H * H_);
    k_embed_gather<<<(T_ - 1) * N_, 128, 0, stream>>>(embed_w, words, x_all);
    k_init<<<(N_ * H_ + 255) / 256, 256, 0, stream>>>(hidden, h, h_bf, acc);

    // sequential recurrence: 4 stream-ordered kernels per step
    for (int t = 0; t < T_ - 1; ++t) {
        k_gru_mm<<<24, 256, 0, stream>>>(x_all, t, h_bf, w_ih_bf, w_hh_bf, gi, gh);
        k_gates<<<(N_ * H_) / 256, 256, 0, stream>>>(gi, gh, b_ih, b_hh, h, h_bf);
        k_emit<<<NCHUNK, 256, 0, stream>>>(h_bf, emit_w_bf, emit_b, words, t, lse_m, lse_s, tgt);
        k_reduce<<<1, 32, 0, stream>>>(lse_m, lse_s, tgt, acc);
    }
    k_out<<<(N_ + N_ * H_ + 255) / 256, 256, 0, stream>>>(acc, h, out);
}
